// TransformerBlock_8581344658039
// MI455X (gfx1250) — matrix operand order repaired and checked by reading, not compiled, not run on hardware
//
#include <hip/hip_runtime.h>
#include <cstddef>

typedef unsigned short ushort_t;
typedef __attribute__((ext_vector_type(16))) __bf16 v16bf;
typedef __attribute__((ext_vector_type(8)))  float  v8f;

union FragAB { v16bf v; uint4 q[2]; };

// ---------------------------------------------------------------- helpers
__device__ __forceinline__ ushort_t f2bf(float f) {
  unsigned u = __float_as_uint(f);
  unsigned r = (u + 0x7FFFu + ((u >> 16) & 1u)) >> 16;  // round-nearest-even
  return (ushort_t)r;
}

// 16-lane crossbar via v_permlane16_b32 (pure VALU, no LDS round-trip).
#if __has_builtin(__builtin_amdgcn_permlane16)
#define HAVE_PERMLANE16 1
#endif

__device__ __forceinline__ float pl16(float v, unsigned s0, unsigned s1) {
#ifdef HAVE_PERMLANE16
  int r = __builtin_amdgcn_permlane16(__float_as_int(v), __float_as_int(v),
                                      (int)s0, (int)s1, false, false);
  return __int_as_float(r);
#else
  (void)s0; (void)s1;
  return v;
#endif
}
// butterfly reductions across the 16 lanes of a half-wave
__device__ __forceinline__ float max16(float v) {
#ifdef HAVE_PERMLANE16
  v = fmaxf(v, pl16(v, 0x67452301u, 0xEFCDAB89u));  // xor 1
  v = fmaxf(v, pl16(v, 0x54761032u, 0xDCFE98BAu));  // xor 2
  v = fmaxf(v, pl16(v, 0x32107654u, 0xBA98FEDCu));  // xor 4
  v = fmaxf(v, pl16(v, 0xFEDCBA98u, 0x76543210u));  // xor 8
#else
#pragma unroll
  for (int off = 1; off < 16; off <<= 1) v = fmaxf(v, __shfl_xor(v, off, 32));
#endif
  return v;
}
__device__ __forceinline__ float sum16(float v) {
#ifdef HAVE_PERMLANE16
  v += pl16(v, 0x67452301u, 0xEFCDAB89u);
  v += pl16(v, 0x54761032u, 0xDCFE98BAu);
  v += pl16(v, 0x32107654u, 0xBA98FEDCu);
  v += pl16(v, 0xFEDCBA98u, 0x76543210u);
#else
#pragma unroll
  for (int off = 1; off < 16; off <<= 1) v += __shfl_xor(v, off, 32);
#endif
  return v;
}

// ---------------------------------------------------------------- LN -> bf16
// one 256-thread block per row, D == 1024
__global__ __launch_bounds__(256) void ln_bf16_kernel(
    const float* __restrict__ x, const float* __restrict__ g,
    const float* __restrict__ b, ushort_t* __restrict__ out) {
  const int D = 1024;
  const int row = blockIdx.x;
  const int t = threadIdx.x;
  const float* xr = x + (size_t)row * D;
  float v[4], s = 0.f, ss = 0.f;
#pragma unroll
  for (int i = 0; i < 4; ++i) { v[i] = xr[t + 256 * i]; s += v[i]; ss += v[i] * v[i]; }
#pragma unroll
  for (int off = 16; off > 0; off >>= 1) {
    s  += __shfl_down(s, off, 32);
    ss += __shfl_down(ss, off, 32);
  }
  __shared__ float red[16];
  const int wid = t >> 5, lane = t & 31;
  if (lane == 0) { red[wid] = s; red[8 + wid] = ss; }
  __syncthreads();
  if (t == 0) {
    float a = 0.f, c = 0.f;
#pragma unroll
    for (int i = 0; i < 8; ++i) { a += red[i]; c += red[8 + i]; }
    red[0] = a; red[8] = c;
  }
  __syncthreads();
  const float mu = red[0] / D;
  const float var = red[8] / D - mu * mu;
  const float rstd = rsqrtf(var + 1e-5f);
  ushort_t* orow = out + (size_t)row * D;
#pragma unroll
  for (int i = 0; i < 4; ++i) {
    int c = t + 256 * i;
    orow[c] = f2bf((v[i] - mu) * rstd * g[c] + b[c]);
  }
}

// ---------------------------------------------------------------- f32 -> bf16 cast
__global__ void cast_bf16_kernel(const float* __restrict__ src,
                                 ushort_t* __restrict__ dst, size_t n) {
  size_t i = (size_t)blockIdx.x * blockDim.x + threadIdx.x;
  size_t stride = (size_t)gridDim.x * blockDim.x;
  for (; i < n; i += stride) dst[i] = f2bf(src[i]);
}

// ---------------------------------------------------------------- bf16 WMMA GEMM
// C[M,N] = A[M,K] @ B[K,N] (+bias / +gelu / +residual*gamma)
// block = 256 thr = 8 waves; block tile 128x128; wave tile 32x64.
// Register double-buffered K loop: loads for step k+1 issue before WMMAs of step k.
// EPI: 0 = bias -> bf16 ; 1 = res + gamma*(acc+bias) -> f32 ; 2 = gelu(acc+bias) -> bf16
struct GTile { FragAB a0, a1, b[4]; };

__device__ __forceinline__ void load_tile(GTile& t, const ushort_t* arow0,
                                          const ushort_t* arow1,
                                          const ushort_t* bcol, int N, int k0) {
  const ushort_t* ap0 = arow0 + k0;
  t.a0.q[0] = *(const uint4*)(ap0);
  t.a0.q[1] = *(const uint4*)(ap0 + 16);
  const ushort_t* ap1 = arow1 + k0;
  t.a1.q[0] = *(const uint4*)(ap1);
  t.a1.q[1] = *(const uint4*)(ap1 + 16);
  const ushort_t* bp = bcol + (size_t)k0 * N;
#pragma unroll
  for (int nf = 0; nf < 4; ++nf) {
#pragma unroll
    for (int s = 0; s < 8; ++s) {
      t.b[nf].v[s] = __builtin_bit_cast(__bf16, bp[(size_t)s * N + nf * 16]);
      t.b[nf].v[8 + s] = __builtin_bit_cast(__bf16, bp[(size_t)(16 + s) * N + nf * 16]);
    }
  }
}
__device__ __forceinline__ void mma_tile(v8f acc[2][4], const GTile& t) {
#pragma unroll
  for (int nf = 0; nf < 4; ++nf) {
    acc[0][nf] = __builtin_amdgcn_wmma_f32_16x16x32_bf16(
        false, t.a0.v, false, t.b[nf].v, (short)0, acc[0][nf], false, false);
    acc[1][nf] = __builtin_amdgcn_wmma_f32_16x16x32_bf16(
        false, t.a1.v, false, t.b[nf].v, (short)0, acc[1][nf], false, false);
  }
}

template <int EPI>
__global__ __launch_bounds__(256) void gemm_bf16_kernel(
    const ushort_t* __restrict__ A, const ushort_t* __restrict__ Bw,
    const float* __restrict__ bias, const float* __restrict__ res,
    const float* __restrict__ gamma, float* __restrict__ outf,
    ushort_t* __restrict__ outb, int M, int N, int K) {
  const int lane = threadIdx.x & 31;
  const int wid  = threadIdx.x >> 5;
  const int wm = wid & 3, wn = wid >> 2;
  const int mtile = blockIdx.y * 128 + wm * 32;
  const int ntile = blockIdx.x * 128 + wn * 64;
  const int hi8 = (lane >> 4) << 3;

  v8f zero = {};
  v8f acc[2][4];
#pragma unroll
  for (int i = 0; i < 2; ++i)
#pragma unroll
    for (int j = 0; j < 4; ++j) acc[i][j] = zero;

  const ushort_t* arow0 = A + (size_t)(mtile + (lane & 15)) * K + hi8;
  const ushort_t* arow1 = arow0 + (size_t)16 * K;
  const ushort_t* bcol  = Bw + (size_t)hi8 * N + ntile + (lane & 15);  // lane&15 = N col, hi8 = K offset (B layout)

  GTile t0, t1;
  load_tile(t0, arow0, arow1, bcol, N, 0);
  for (int k0 = 0; k0 < K; k0 += 64) {
    load_tile(t1, arow0, arow1, bcol, N, k0 + 32);   // prefetch odd step
    mma_tile(acc, t0);
    if (k0 + 64 < K) load_tile(t0, arow0, arow1, bcol, N, k0 + 64);  // prefetch even
    mma_tile(acc, t1);
  }

#pragma unroll
  for (int nf = 0; nf < 4; ++nf) {
    const int n = ntile + nf * 16 + (lane & 15);
    const float bn = bias[n];
    const float gn = (EPI == 1) ? gamma[n] : 0.f;
#pragma unroll
    for (int mf = 0; mf < 2; ++mf)
#pragma unroll
      for (int r = 0; r < 8; ++r) {
        const int m = mtile + mf * 16 + r + hi8;
        const size_t idx = (size_t)m * N + n;
        float v = acc[mf][nf][r] + bn;
        if (EPI == 0) {
          outb[idx] = f2bf(v);
        } else if (EPI == 1) {
          outf[idx] = res[idx] + gn * v;
        } else {
          float ge = 0.5f * v * (1.f + erff(v * 0.70710678f));  // exact GELU
          outb[idx] = f2bf(ge);
        }
      }
  }
}

// ---------------------------------------------------------------- K transpose
// qkv[B*N,3072] (K at col 1024+head*64+d)  ->  kT[bh][d][n]  (bf16)
__global__ void kTranspose_kernel(const ushort_t* __restrict__ qkv,
                                  ushort_t* __restrict__ kT) {
  unsigned idx = blockIdx.x * blockDim.x + threadIdx.x;  // < 128*64*1024
  const int n = idx & 1023;
  const int d = (idx >> 10) & 63;
  const int bh = idx >> 16;
  const int b = bh >> 4, head = bh & 15;
  kT[idx] = qkv[(size_t)(b * 1024 + n) * 3072 + 1024 + head * 64 + d];
}

// ---------------------------------------------------------------- attention
// grid (N/128, B*H); block 256 = 8 waves; one wave = 16 Q rows, flash-style.
// K-fragments register double-buffered across kv tiles; V loads issued early.
__global__ __launch_bounds__(256) void attn_kernel(
    const ushort_t* __restrict__ qkv, const ushort_t* __restrict__ kT,
    ushort_t* __restrict__ attnO) {
  __shared__ __align__(16) ushort_t pbuf[8 * 16 * 32];  // P tile per wave (bf16)
  const int lane = threadIdx.x & 31;
  const int w = threadIdx.x >> 5;
  const int bh = blockIdx.y;
  const int b = bh >> 4, head = bh & 15;
  const int q0 = blockIdx.x * 128 + w * 16;
  const int hi8 = (lane >> 4) << 3;
  const float scale = 0.125f;  // 1/sqrt(64)
  const int brow = b * 1024;

  // Q: 16x64 -> two 16x32 A-fragments, register-resident
  FragAB qf[2];
  {
    const ushort_t* qp = qkv + (size_t)(brow + q0 + (lane & 15)) * 3072 +
                         head * 64 + hi8;
    qf[0].q[0] = *(const uint4*)(qp);
    qf[0].q[1] = *(const uint4*)(qp + 16);
    qf[1].q[0] = *(const uint4*)(qp + 32);
    qf[1].q[1] = *(const uint4*)(qp + 48);
  }

  v8f zero = {};
  v8f o[4];
#pragma unroll
  for (int c = 0; c < 4; ++c) o[c] = zero;
  float rmax[8], rsum[8];
#pragma unroll
  for (int r = 0; r < 8; ++r) { rmax[r] = -1e30f; rsum[r] = 0.f; }

  ushort_t* pw = pbuf + w * 512;
  const ushort_t* kbase = kT + (size_t)bh * 64 * 1024;

  // K B-fragment loader: bk[nc][kk] covers kv cols j0+nc*16.. , hd rows kk*32..
  auto loadK = [&](FragAB (&dst)[2][2], int j0) {
#pragma unroll
    for (int nc = 0; nc < 2; ++nc) {
      const ushort_t* kp0 = kbase + (size_t)hi8 * 1024 + j0 + nc * 16 + (lane & 15);
      const ushort_t* kp1 = kp0 + (size_t)32 * 1024;
#pragma unroll
      for (int s = 0; s < 8; ++s) {
        dst[nc][0].v[s] = __builtin_bit_cast(__bf16, kp0[(size_t)s * 1024]);
        dst[nc][0].v[8 + s] = __builtin_bit_cast(__bf16, kp0[(size_t)(16 + s) * 1024]);
        dst[nc][1].v[s] = __builtin_bit_cast(__bf16, kp1[(size_t)s * 1024]);
        dst[nc][1].v[8 + s] = __builtin_bit_cast(__bf16, kp1[(size_t)(16 + s) * 1024]);
      }
    }
  };

  // one 32-kv-row flash step using pre-loaded K fragments
  auto step = [&](int j0, FragAB (&bk)[2][2]) {
    // V loads issued first: latency hides under S-WMMA + softmax VALU
    FragAB vf[4];
    const ushort_t* vp =
        qkv + (size_t)(brow + j0 + hi8) * 3072 + 2048 + head * 64 + (lane & 15);
#pragma unroll
    for (int c = 0; c < 4; ++c) {
#pragma unroll
      for (int s = 0; s < 8; ++s) {
        vf[c].v[s] = __builtin_bit_cast(__bf16, vp[(size_t)s * 3072 + c * 16]);
        vf[c].v[8 + s] = __builtin_bit_cast(__bf16, vp[(size_t)(16 + s) * 3072 + c * 16]);
      }
    }
    // S = Q @ K^T -> two 16x16 f32 tiles
    v8f s[2];
    s[0] = zero; s[1] = zero;
#pragma unroll
    for (int nc = 0; nc < 2; ++nc) {
      s[nc] = __builtin_amdgcn_wmma_f32_16x16x32_bf16(
          false, qf[0].v, false, bk[nc][0].v, (short)0, s[nc], false, false);
      s[nc] = __builtin_amdgcn_wmma_f32_16x16x32_bf16(
          false, qf[1].v, false, bk[nc][1].v, (short)0, s[nc], false, false);
    }
    // online softmax; row reductions via v_permlane16 (no LDS)
    float corr[8];
#pragma unroll
    for (int r = 0; r < 8; ++r) {
      float s0 = s[0][r] * scale;
      float s1 = s[1][r] * scale;
      float tmax = max16(fmaxf(s0, s1));
      float mnew = fmaxf(rmax[r], tmax);
      float c = __expf(rmax[r] - mnew);
      float p0 = __expf(s0 - mnew);
      float p1 = __expf(s1 - mnew);
      float ts = sum16(p0 + p1);
      rsum[r] = rsum[r] * c + ts;
      rmax[r] = mnew;
      corr[r] = c;
      const int m = r + hi8;
      pw[m * 32 + (lane & 15)] = f2bf(p0);
      pw[m * 32 + 16 + (lane & 15)] = f2bf(p1);
    }
#pragma unroll
    for (int c = 0; c < 4; ++c)
#pragma unroll
      for (int r = 0; r < 8; ++r) o[c][r] *= corr[r];

    asm volatile("s_wait_dscnt 0" ::: "memory");  // LDS write -> read, same wave

    FragAB pf;  // P as 16x32 A-fragment from LDS
    {
      const ushort_t* pp = pw + (lane & 15) * 32 + hi8;
      pf.q[0] = *(const uint4*)(pp);
      pf.q[1] = *(const uint4*)(pp + 16);
    }
#pragma unroll
    for (int c = 0; c < 4; ++c)
      o[c] = __builtin_amdgcn_wmma_f32_16x16x32_bf16(
          false, pf.v, false, vf[c].v, (short)0, o[c], false, false);
  };

  FragAB bkA[2][2], bkB[2][2];
  loadK(bkA, 0);
  for (int j0 = 0; j0 < 1024; j0 += 64) {
    loadK(bkB, j0 + 32);                    // prefetch odd tile
    step(j0, bkA);
    if (j0 + 64 < 1024) loadK(bkA, j0 + 64);  // prefetch even tile
    step(j0 + 32, bkB);
  }

#pragma unroll
  for (int c = 0; c < 4; ++c)
#pragma unroll
    for (int r = 0; r < 8; ++r) {
      const int m = r + hi8;
      float v = o[c][r] / rsum[r];
      attnO[(size_t)(brow + q0 + m) * 1024 + head * 64 + c * 16 +
            (lane & 15)] = f2bf(v);
    }
}

// ---------------------------------------------------------------- launch
extern "C" void kernel_launch(void* const* d_in, const int* in_sizes, int n_in,
                              void* d_out, int out_size, void* d_ws, size_t ws_size,
                              hipStream_t stream) {
  const float* x      = (const float*)d_in[0];
  const float* ln1_g  = (const float*)d_in[1];
  const float* ln1_b  = (const float*)d_in[2];
  const float* qkv_w  = (const float*)d_in[3];
  const float* qkv_b  = (const float*)d_in[4];
  const float* proj_w = (const float*)d_in[5];
  const float* proj_b = (const float*)d_in[6];
  const float* ls1_g  = (const float*)d_in[7];
  const float* ln2_g  = (const float*)d_in[8];
  const float* ln2_b  = (const float*)d_in[9];
  const float* fc1_w  = (const float*)d_in[10];
  const float* fc1_b  = (const float*)d_in[11];
  const float* fc2_w  = (const float*)d_in[12];
  const float* fc2_b  = (const float*)d_in[13];
  const float* ls2_g  = (const float*)d_in[14];
  float* out = (float*)d_out;

  const int Bz = 8, N = 1024, D = 1024, H = 16, HID = 4096;
  const int R = Bz * N;          // 8192 rows
  const int QN = 3 * D;          // 3072

  char* w = (char*)d_ws;
  size_t off = 0;
  auto alloc = [&](size_t bytes) -> void* {
    void* p = w + off;
    off = (off + bytes + 255) & ~(size_t)255;
    return p;
  };
  ushort_t* hbf   = (ushort_t*)alloc((size_t)R * D * 2);     // LN1 out (bf16)
  ushort_t* wqkv  = (ushort_t*)alloc((size_t)D * QN * 2);
  ushort_t* wproj = (ushort_t*)alloc((size_t)D * D * 2);
  ushort_t* wfc1  = (ushort_t*)alloc((size_t)D * HID * 2);
  ushort_t* wfc2  = (ushort_t*)alloc((size_t)HID * D * 2);
  ushort_t* qkv   = (ushort_t*)alloc((size_t)R * QN * 2);    // bf16
  ushort_t* kT    = (ushort_t*)alloc((size_t)Bz * H * 64 * N * 2);
  ushort_t* attnO = (ushort_t*)alloc((size_t)R * D * 2);
  float*    x1    = (float*)alloc((size_t)R * D * 4);        // post-attn residual
  ushort_t* h2ln  = (ushort_t*)alloc((size_t)R * D * 2);     // LN2 out (bf16)
  ushort_t* h2    = (ushort_t*)alloc((size_t)R * HID * 2);   // GELU(fc1) (bf16)
  (void)ws_size; (void)in_sizes; (void)n_in; (void)out_size;

  // 1) LN1
  ln_bf16_kernel<<<R, 256, 0, stream>>>(x, ln1_g, ln1_b, hbf);
  // 2) weight casts (bf16)
  cast_bf16_kernel<<<4096, 256, 0, stream>>>(qkv_w, wqkv, (size_t)D * QN);
  cast_bf16_kernel<<<4096, 256, 0, stream>>>(proj_w, wproj, (size_t)D * D);
  cast_bf16_kernel<<<4096, 256, 0, stream>>>(fc1_w, wfc1, (size_t)D * HID);
  cast_bf16_kernel<<<4096, 256, 0, stream>>>(fc2_w, wfc2, (size_t)HID * D);
  // 3) QKV = LN1(x) @ qkv_w + b   -> bf16
  gemm_bf16_kernel<0><<<dim3(QN / 128, R / 128), 256, 0, stream>>>(
      hbf, wqkv, qkv_b, nullptr, nullptr, nullptr, qkv, R, QN, D);
  // 4) transpose K for contiguous B-fragments
  kTranspose_kernel<<<(Bz * H * 64 * N) / 256, 256, 0, stream>>>(qkv, kT);
  // 5) attention (flash-style, WMMA)
  attn_kernel<<<dim3(N / 128, Bz * H), 256, 0, stream>>>(qkv, kT, attnO);
  // 6) x1 = x + ls1 * (attnO @ proj_w + b)
  gemm_bf16_kernel<1><<<dim3(D / 128, R / 128), 256, 0, stream>>>(
      attnO, wproj, proj_b, x, ls1_g, x1, nullptr, R, D, D);
  // 7) LN2
  ln_bf16_kernel<<<R, 256, 0, stream>>>(x1, ln2_g, ln2_b, h2ln);
  // 8) h2 = gelu(LN2 @ fc1_w + b)  -> bf16
  gemm_bf16_kernel<2><<<dim3(HID / 128, R / 128), 256, 0, stream>>>(
      h2ln, wfc1, fc1_b, nullptr, nullptr, nullptr, h2, R, HID, D);
  // 9) out = x1 + ls2 * (h2 @ fc2_w + b)
  gemm_bf16_kernel<1><<<dim3(D / 128, R / 128), 256, 0, stream>>>(
      h2, wfc2, fc2_b, x1, ls2_g, out, nullptr, R, D, HID);
}
